// HadamardQuantLinear_21552145891805
// MI455X (gfx1250) — 8-bit matrix operand order repaired and checked by reading, compiled, not run on hardware
//
#include <hip/hip_runtime.h>
#include <stdint.h>

#define K_DIM 4096
#define HAD   64

typedef __attribute__((ext_vector_type(16))) int   v16i;
typedef __attribute__((ext_vector_type(8)))  float v8f;

__device__ __forceinline__ unsigned fp8_e4m3_sw(float v) {
  unsigned b    = __float_as_uint(v);
  unsigned sign = (b >> 24) & 0x80u;
  unsigned ab   = b & 0x7fffffffu;
  if (ab > 0x43e00000u) ab = 0x43e00000u;
  int e = (int)(ab >> 23) - 127;
  unsigned enc;
  if (e < -6) {
    float av = __uint_as_float(ab);
    enc = (unsigned)__float2int_rn(av * 512.0f);
  } else {
    unsigned lsb = (ab >> 20) & 1u;
    ab += 0x0007FFFFu + lsb;
    unsigned mant = (ab >> 20) & 7u;
    int ee = (int)(ab >> 23) - 127;
    if (ee > 8) { ee = 8; mant = 7u; }
    enc = ((unsigned)(ee + 7) << 3) | mant;
  }
  return sign | enc;
}

__device__ __forceinline__ unsigned pack4_fp8(float a, float b, float c, float d) {
#if __has_builtin(__builtin_amdgcn_cvt_pk_fp8_f32)
  int w = __builtin_amdgcn_cvt_pk_fp8_f32(a, b, 0, false);
  w     = __builtin_amdgcn_cvt_pk_fp8_f32(c, d, w, true);
  return (unsigned)w;
#else
  return fp8_e4m3_sw(a) | (fp8_e4m3_sw(b) << 8) |
         (fp8_e4m3_sw(c) << 16) | (fp8_e4m3_sw(d) << 24);
#endif
}

__global__ void __launch_bounds__(256) hq_rotquant_kernel(
    const float* __restrict__ X, const float* __restrict__ H,
    unsigned char* __restrict__ Q, float* __restrict__ S) {
  __shared__ float ldsH[HAD * HAD];
  __shared__ float ldsX[HAD * 68];
  __shared__ float ldsRed[8];

  const int t   = threadIdx.x;
  const size_t row = blockIdx.x;
  const int b   = t >> 2;
  const int jb  = (t & 3) << 4;

  {
    const float4* hs = (const float4*)H;
    float4*       hd = (float4*)ldsH;
#pragma unroll
    for (int i = 0; i < 4; ++i) hd[t * 4 + i] = hs[t * 4 + i];

    const float4* xs = (const float4*)(X + row * K_DIM + t * 16);
    float4*       xd = (float4*)&ldsX[b * 68 + jb];
#pragma unroll
    for (int i = 0; i < 4; ++i) xd[i] = xs[i];
  }
  __syncthreads();

  float acc[16];
#pragma unroll
  for (int i = 0; i < 16; ++i) acc[i] = 0.0f;
  const float* xb = &ldsX[b * 68];
#pragma unroll 4
  for (int k = 0; k < HAD; ++k) {
    const float xk = xb[k];
    const float4* hr = (const float4*)&ldsH[k * HAD + jb];
    const float4 h0 = hr[0], h1 = hr[1], h2 = hr[2], h3 = hr[3];
    acc[0]  = fmaf(xk, h0.x, acc[0]);  acc[1]  = fmaf(xk, h0.y, acc[1]);
    acc[2]  = fmaf(xk, h0.z, acc[2]);  acc[3]  = fmaf(xk, h0.w, acc[3]);
    acc[4]  = fmaf(xk, h1.x, acc[4]);  acc[5]  = fmaf(xk, h1.y, acc[5]);
    acc[6]  = fmaf(xk, h1.z, acc[6]);  acc[7]  = fmaf(xk, h1.w, acc[7]);
    acc[8]  = fmaf(xk, h2.x, acc[8]);  acc[9]  = fmaf(xk, h2.y, acc[9]);
    acc[10] = fmaf(xk, h2.z, acc[10]); acc[11] = fmaf(xk, h2.w, acc[11]);
    acc[12] = fmaf(xk, h3.x, acc[12]); acc[13] = fmaf(xk, h3.y, acc[13]);
    acc[14] = fmaf(xk, h3.z, acc[14]); acc[15] = fmaf(xk, h3.w, acc[15]);
  }

  float m = 0.0f;
#pragma unroll
  for (int i = 0; i < 16; ++i) m = fmaxf(m, fabsf(acc[i]));
#pragma unroll
  for (int off = 16; off; off >>= 1) m = fmaxf(m, __shfl_xor(m, off, 32));
  if ((t & 31) == 0) ldsRed[t >> 5] = m;
  __syncthreads();
  if (t == 0) {
    float mm = ldsRed[0];
#pragma unroll
    for (int w = 1; w < 8; ++w) mm = fmaxf(mm, ldsRed[w]);
    const float s = fmaxf(mm * (1.0f / 448.0f), 1e-12f);
    S[row]    = s;
    ldsRed[0] = 1.0f / s;
  }
  __syncthreads();
  const float inv = ldsRed[0];

  uint4 pk;
  pk.x = pack4_fp8(acc[0]  * inv, acc[1]  * inv, acc[2]  * inv, acc[3]  * inv);
  pk.y = pack4_fp8(acc[4]  * inv, acc[5]  * inv, acc[6]  * inv, acc[7]  * inv);
  pk.z = pack4_fp8(acc[8]  * inv, acc[9]  * inv, acc[10] * inv, acc[11] * inv);
  pk.w = pack4_fp8(acc[12] * inv, acc[13] * inv, acc[14] * inv, acc[15] * inv);
  *(uint4*)(Q + row * K_DIM + t * 16) = pk;
}

#define WG_M   128
#define WG_N   256
#define KSTEP  128
#define ABYTES (WG_M * KSTEP)
#define BBYTES (WG_N * KSTEP)
#define BUFSZ  (ABYTES + BBYTES)

__device__ __forceinline__ void async_copy16(uint32_t ldsAddr,
                                             const unsigned char* g) {
  asm volatile("global_load_async_to_lds_b128 %0, %1, off"
               :: "v"(ldsAddr), "v"(g)
               : "memory");
}
__device__ __forceinline__ void async_copy8(uint32_t ldsAddr,
                                            const unsigned char* g) {
  asm volatile("global_load_async_to_lds_b64 %0, %1, off"
               :: "v"(ldsAddr), "v"(g)
               : "memory");
}

__global__ void __launch_bounds__(256) hq_gemm_fp8_kernel(
    const unsigned char* __restrict__ Aq, const unsigned char* __restrict__ Bq,
    const float* __restrict__ sA, const float* __restrict__ sB,
    const float* __restrict__ bias, float* __restrict__ Out,
    int M, int N) {
  __shared__ unsigned char ldsBuf[2][BUFSZ];

  const int t    = threadIdx.x;
  const int lane = t & 31;
  const int wave = t >> 5;
  const int wm   = wave & 1;
  const int wn   = wave >> 1;
  const int l16  = lane & 15;
  const int lh   = lane >> 4;
  const int mBase = blockIdx.y * WG_M;
  const int nBase = blockIdx.x * WG_N;

  const int prow = t >> 3;
  const int pcid = t & 7;
  const unsigned char* aG = Aq + (size_t)(mBase + prow) * K_DIM + pcid * 16;
  const unsigned char* bG = Bq + (size_t)(nBase + prow) * K_DIM + pcid * 16;
  const uint32_t ldsBase = (uint32_t)(uintptr_t)&ldsBuf[0][0];
  const uint32_t aL = ldsBase + (uint32_t)(prow * KSTEP) +
                      (uint32_t)((((pcid >> 1) ^ ((prow >> 1) & 3)) << 5) +
                                 ((pcid & 1) << 4));
  const uint32_t bL = ldsBase + ABYTES + (uint32_t)(prow * KSTEP) +
                      (uint32_t)(((pcid ^ (prow & 7)) << 4));

  v8f acc[4][4];
#pragma unroll
  for (int mi = 0; mi < 4; ++mi)
#pragma unroll
    for (int ni = 0; ni < 4; ++ni) { v8f z = {}; acc[mi][ni] = z; }

#pragma unroll
  for (int i = 0; i < 4; ++i) {
    async_copy8(aL + (uint32_t)i * 4096u,       aG + (size_t)i * 32 * K_DIM);
    async_copy8(aL + (uint32_t)i * 4096u + 8u, aG + (size_t)i * 32 * K_DIM + 8);
  }
#pragma unroll
  for (int i = 0; i < 8; ++i)
    async_copy16(bL + (uint32_t)i * 4096u, bG + (size_t)i * 32 * K_DIM);

#pragma unroll 1
  for (int ki = 0; ki < K_DIM / KSTEP; ++ki) {
    const int cur = ki & 1;
    if (ki < K_DIM / KSTEP - 1) {
      const uint32_t bs = (uint32_t)((cur ^ 1) * BUFSZ);
      const int kb = (ki + 1) * KSTEP;
#pragma unroll
      for (int i = 0; i < 4; ++i) {
        async_copy8(aL + bs + (uint32_t)i * 4096u,
                    aG + kb + (size_t)i * 32 * K_DIM);
        async_copy8(aL + bs + (uint32_t)i * 4096u + 8u,
                    aG + kb + (size_t)i * 32 * K_DIM + 8);
      }
#pragma unroll
      for (int i = 0; i < 8; ++i)
        async_copy16(bL + bs + (uint32_t)i * 4096u,
                     bG + kb + (size_t)i * 32 * K_DIM);
      asm volatile("s_wait_asynccnt 16" ::: "memory");
    } else {
      asm volatile("s_wait_asynccnt 0" ::: "memory");
    }
    __syncthreads();

    const unsigned char* aBuf = &ldsBuf[cur][0];
    const unsigned char* bBuf = &ldsBuf[cur][ABYTES];

    v16i bf[4];
#pragma unroll
    for (int ni = 0; ni < 4; ++ni) {
      const int n = wn * 64 + ni * 16 + l16;
      const unsigned char* bcol = bBuf + n * KSTEP;
      const int swn = (n & 7) << 4;
#pragma unroll
      for (int g = 0; g < 4; ++g) {
        const int cid = 2 * g + lh;
        const uint4 d = *(const uint4*)(bcol + (((cid << 4) ^ swn)));
        bf[ni][4 * g]     = (int)d.x;
        bf[ni][4 * g + 1] = (int)d.y;
        bf[ni][4 * g + 2] = (int)d.z;
        bf[ni][4 * g + 3] = (int)d.w;
      }
    }

#pragma unroll
    for (int mi = 0; mi < 4; ++mi) {
      const int r = wm * 64 + mi * 16 + l16;
      const unsigned char* arow = aBuf + r * KSTEP + lh * 16;
      const int swq = (r >> 1) & 3;
      v16i af;
#pragma unroll
      for (int q = 0; q < 4; ++q) {
        const uint4 d = *(const uint4*)(arow + ((q ^ swq) << 5));
        af[4 * q]     = (int)d.x;
        af[4 * q + 1] = (int)d.y;
        af[4 * q + 2] = (int)d.z;
        af[4 * q + 3] = (int)d.w;
      }
#pragma unroll
      for (int ni = 0; ni < 4; ++ni)
        acc[mi][ni] = __builtin_amdgcn_wmma_f32_16x16x128_fp8_fp8(
            af, bf[ni], (short)0, acc[mi][ni], false, false);
    }
    __syncthreads();
  }

#pragma unroll
  for (int ni = 0; ni < 4; ++ni) {
    const int n    = nBase + wn * 64 + ni * 16 + l16;
    const float sn = sB[n];
    const float bn = bias[n];
#pragma unroll
    for (int mi = 0; mi < 4; ++mi) {
      const int m0 = mBase + wm * 64 + mi * 16 + lh * 8;
#pragma unroll
      for (int r = 0; r < 8; ++r) {
        const int m = m0 + r;
        Out[(size_t)m * N + n] = acc[mi][ni][r] * (sA[m] * sn) + bn;
      }
    }
  }
}

extern "C" void kernel_launch(void* const* d_in, const int* in_sizes, int n_in,
                              void* d_out, int out_size, void* d_ws, size_t ws_size,
                              hipStream_t stream) {
  const float* x    = (const float*)d_in[0];
  const float* wgt  = (const float*)d_in[1];
  const float* bias = (const float*)d_in[2];
  const float* H    = (const float*)d_in[3];
  float* out = (float*)d_out;

  const int N = in_sizes[2];
  const int K = in_sizes[1] / N;
  const int M = in_sizes[0] / K;

  unsigned char* xq = (unsigned char*)d_ws;
  unsigned char* wq = xq + (size_t)M * K;
  float* sx = (float*)(wq + (size_t)N * K);
  float* sw = sx + M;

  hq_rotquant_kernel<<<N, 256, 0, stream>>>(wgt, H, wq, sw);
  hq_rotquant_kernel<<<M, 256, 0, stream>>>(x,   H, xq, sx);

  dim3 grid(N / WG_N, M / WG_M);
  hq_gemm_fp8_kernel<<<grid, 256, 0, stream>>>(xq, wq, sx, sw, bias, out, M, N);
}
